// OmegaMoELayer_65773129171181
// MI455X (gfx1250) — matrix operand order repaired and checked by reading, not compiled, not run on hardware
//
#include <hip/hip_runtime.h>
#include <hip/hip_bf16.h>
#include <stdint.h>

typedef __attribute__((ext_vector_type(16))) __bf16 v16bf;
typedef __attribute__((ext_vector_type(8)))  float  v8f;

union FragAB {
    uint4 q[2];
    v16bf v;
};

__device__ __forceinline__ unsigned short f2bf(float f) {
    unsigned int u = __float_as_uint(f);
    u += 0x7FFFu + ((u >> 16) & 1u);          // round-to-nearest-even
    return (unsigned short)(u >> 16);
}
__device__ __forceinline__ unsigned int packbf2(float lo, float hi) {
    return (unsigned int)f2bf(lo) | ((unsigned int)f2bf(hi) << 16);
}
__device__ __forceinline__ float relu2(float g) {
    float r = g > 0.f ? g : 0.f;
    return r * r;
}

// ---------------------------------------------------------------------------
// Kernel 1: x fp32 -> bf16 (row-major, coalesced); also zero expert counters.
// ---------------------------------------------------------------------------
__global__ __launch_bounds__(256) void convert_x_kernel(const float* __restrict__ x,
                                                        uint2* __restrict__ xb2,
                                                        int* __restrict__ counts, int n4) {
    int gid = blockIdx.x * blockDim.x + threadIdx.x;
    if (gid < 8) counts[gid] = 0;
    int stride = gridDim.x * blockDim.x;
    for (int i = gid; i < n4; i += stride) {
        float4 v = ((const float4*)x)[i];
        uint2 p;
        p.x = packbf2(v.x, v.y);
        p.y = packbf2(v.z, v.w);
        xb2[i] = p;
    }
}

// ---------------------------------------------------------------------------
// Kernel 2: pack an [E][K][N] fp32 weight into bf16 WMMA B-fragment order.
// Tile = 32(K) x 16(N). Per tile: 32 lanes x 8 dwords (2 bf16/dword, K-pairs),
// lane l: N = nt*16 + (l&15), dword j: K = (l>>4)*8 + (j&3)*2 + (j>>2)*16. One dword per thread.
// ---------------------------------------------------------------------------
__global__ __launch_bounds__(256) void pack_w_kernel(const float* __restrict__ src,
                                                     unsigned int* __restrict__ dst,
                                                     int Kdim, int Ndim) {
    int tilesN = Ndim >> 4;
    int tilesK = Kdim >> 5;
    int perExpert = tilesK * tilesN;
    int total = 8 * perExpert * 256;
    int tid = blockIdx.x * blockDim.x + threadIdx.x;
    if (tid >= total) return;
    int j      = tid & 7;
    int l      = (tid >> 3) & 31;
    int rest   = tid >> 8;
    int tileId = rest % perExpert;
    int e      = rest / perExpert;
    int kt = tileId / tilesN, nt = tileId % tilesN;
    int K0 = kt * 32 + ((l >> 4) << 3) + ((j & 3) << 1) + ((j >> 2) << 4);
    int N  = nt * 16 + (l & 15);
    size_t sbase = (size_t)e * Kdim * Ndim;
    float s0 = src[sbase + (size_t)K0 * Ndim + N];
    float s1 = src[sbase + (size_t)(K0 + 1) * Ndim + N];
    dst[((size_t)e * perExpert + tileId) * 256 + l * 8 + j] = packbf2(s0, s1);
}

// ---------------------------------------------------------------------------
// Kernel 3: router. gate_w (8x1024 fp32 = 32KB) in LDS; one token per thread;
// argmax (first-max tie-break == jnp.argmax); atomic scatter into buckets.
// ---------------------------------------------------------------------------
__global__ __launch_bounds__(256) void router_kernel(const float* __restrict__ x,
                                                     const float* __restrict__ gate_w,
                                                     int* __restrict__ bucket,
                                                     int* __restrict__ counts) {
    __shared__ float gw[8 * 1024];
    int tid = threadIdx.x;
    for (int i = tid; i < 8 * 1024; i += 256) gw[i] = gate_w[i];
    __syncthreads();

    int t = blockIdx.x * 256 + tid;
    const float4* xr = (const float4*)(x + (size_t)t * 1024);
    float acc[8] = {};
    for (int h4 = 0; h4 < 256; ++h4) {
        float4 v = xr[h4];
        #pragma unroll
        for (int e = 0; e < 8; ++e) {
            const float* g = &gw[e * 1024 + h4 * 4];
            acc[e] += v.x * g[0] + v.y * g[1] + v.z * g[2] + v.w * g[3];
        }
    }
    int best = 0; float bv = acc[0];
    #pragma unroll
    for (int e = 1; e < 8; ++e)
        if (acc[e] > bv) { bv = acc[e]; best = e; }
    int pos = atomicAdd(&counts[best], 1);
    bucket[best * 16384 + pos] = t;
}

// ---------------------------------------------------------------------------
// Kernel 4: expert FFN, M = 32 tokens per block. grid (E, T/32); 8 waves.
// Each B fragment is shared by two 16-row M-tiles -> 2x arithmetic intensity
// on the L2 weight stream vs an M=16 tiling.
// ---------------------------------------------------------------------------
__global__ __launch_bounds__(256) void moe_ffn_kernel(
        const unsigned short* __restrict__ xb,
        const unsigned int*  __restrict__ wgp,
        const unsigned int*  __restrict__ wup,
        const unsigned int*  __restrict__ wdp,
        const int* __restrict__ bucket,
        const int* __restrict__ counts,
        float* __restrict__ out) {
    constexpr int H = 1024, D = 256, T = 16384;
    int e    = blockIdx.x;
    int tile = blockIdx.y;
    int cnt  = counts[e];
    int base = tile * 32;
    if (base >= cnt) return;                       // block-uniform early exit
    int ntok = cnt - base; if (ntok > 32) ntok = 32;

    __shared__ __align__(16) unsigned short xt[32][1024 + 8];  // 66KB, pad kills bank conflicts
    __shared__ __align__(16) unsigned short it[32][256 + 8];   // 17KB
    __shared__ int toks[32];

    int tid = threadIdx.x;
    if (tid < 32) {
        int idx = tid < ntok ? tid : (ntok - 1);   // clamp: duplicated rows, stores masked
        toks[tid] = bucket[e * T + base + idx];
    }
    __syncthreads();

    {   // stage 32 x-rows (bf16) into LDS: 8 threads/row, 16 x 16B chunks each
        int row = tid >> 3;
        int c0  = (tid & 7) * 128;
        const uint4* src = (const uint4*)(xb + (size_t)toks[row] * H + c0);
        uint4* dst = (uint4*)&xt[row][c0];
        #pragma unroll
        for (int j = 0; j < 16; ++j) dst[j] = src[j];
    }
    __syncthreads();

    int lane = tid & 31, wave = tid >> 5;
    int sel  = lane >> 4;          // K-half selector (A layout)
    int arow = lane & 15;

    // accumulators: [M-half 0/1][D-tile 0/1] for gate and up
    v8f gA0 = {}, gA1 = {}, gB0 = {}, gB1 = {};
    v8f uA0 = {}, uA1 = {}, uB0 = {}, uB1 = {};
    size_t ebase = (size_t)e * (H * D / 2);        // packed dwords per expert
    int nt0 = wave * 2;                            // this wave's D-tile pair (of 16)

    for (int kt = 0; kt < H / 32; ++kt) {
        int k0 = kt * 32;
        FragAB A0, A1;                             // rows 0-15 and 16-31
        A0.q[0] = *(const uint4*)&xt[arow][k0 + sel * 8];
        A0.q[1] = *(const uint4*)&xt[arow][k0 + 16 + sel * 8];
        A1.q[0] = *(const uint4*)&xt[arow + 16][k0 + sel * 8];
        A1.q[1] = *(const uint4*)&xt[arow + 16][k0 + 16 + sel * 8];

        const uint4* bg0 = (const uint4*)(wgp + ebase + ((size_t)(kt * 16 + nt0)     * 256 + lane * 8));
        const uint4* bg1 = (const uint4*)(wgp + ebase + ((size_t)(kt * 16 + nt0 + 1) * 256 + lane * 8));
        const uint4* bu0 = (const uint4*)(wup + ebase + ((size_t)(kt * 16 + nt0)     * 256 + lane * 8));
        const uint4* bu1 = (const uint4*)(wup + ebase + ((size_t)(kt * 16 + nt0 + 1) * 256 + lane * 8));
        FragAB B0, B1, B2, B3;
        B0.q[0] = bg0[0]; B0.q[1] = bg0[1];
        B1.q[0] = bg1[0]; B1.q[1] = bg1[1];
        B2.q[0] = bu0[0]; B2.q[1] = bu0[1];
        B3.q[0] = bu1[0]; B3.q[1] = bu1[1];

        gA0 = __builtin_amdgcn_wmma_f32_16x16x32_bf16(false, A0.v, false, B0.v, (short)0, gA0, false, false);
        gB0 = __builtin_amdgcn_wmma_f32_16x16x32_bf16(false, A1.v, false, B0.v, (short)0, gB0, false, false);
        gA1 = __builtin_amdgcn_wmma_f32_16x16x32_bf16(false, A0.v, false, B1.v, (short)0, gA1, false, false);
        gB1 = __builtin_amdgcn_wmma_f32_16x16x32_bf16(false, A1.v, false, B1.v, (short)0, gB1, false, false);
        uA0 = __builtin_amdgcn_wmma_f32_16x16x32_bf16(false, A0.v, false, B2.v, (short)0, uA0, false, false);
        uB0 = __builtin_amdgcn_wmma_f32_16x16x32_bf16(false, A1.v, false, B2.v, (short)0, uB0, false, false);
        uA1 = __builtin_amdgcn_wmma_f32_16x16x32_bf16(false, A0.v, false, B3.v, (short)0, uA1, false, false);
        uB1 = __builtin_amdgcn_wmma_f32_16x16x32_bf16(false, A1.v, false, B3.v, (short)0, uB1, false, false);
    }

    {   // inter = relu(g)^2 * u, in C layout (g and u share it) -> bf16 LDS tile
        int ncol = lane & 15;
        #pragma unroll
        for (int r = 0; r < 8; ++r) {
            int m = r + sel * 8;
            it[m][wave * 32 + ncol]           = f2bf(relu2(gA0[r]) * uA0[r]);
            it[m][wave * 32 + 16 + ncol]      = f2bf(relu2(gA1[r]) * uA1[r]);
            it[m + 16][wave * 32 + ncol]      = f2bf(relu2(gB0[r]) * uB0[r]);
            it[m + 16][wave * 32 + 16 + ncol] = f2bf(relu2(gB1[r]) * uB1[r]);
        }
    }
    __syncthreads();

    size_t dbase = (size_t)e * (D * H / 2);
    for (int nn = 0; nn < 8; ++nn) {
        int nt = wave * 8 + nn;                    // H-tile index (of 64)
        v8f accA = {}, accB = {};
        #pragma unroll
        for (int kt = 0; kt < D / 32; ++kt) {
            int k0 = kt * 32;
            FragAB A0, A1;
            A0.q[0] = *(const uint4*)&it[arow][k0 + sel * 8];
            A0.q[1] = *(const uint4*)&it[arow][k0 + 16 + sel * 8];
            A1.q[0] = *(const uint4*)&it[arow + 16][k0 + sel * 8];
            A1.q[1] = *(const uint4*)&it[arow + 16][k0 + 16 + sel * 8];
            const uint4* bp = (const uint4*)(wdp + dbase + ((size_t)(kt * 64 + nt) * 256 + lane * 8));
            FragAB Bf; Bf.q[0] = bp[0]; Bf.q[1] = bp[1];
            accA = __builtin_amdgcn_wmma_f32_16x16x32_bf16(false, A0.v, false, Bf.v, (short)0, accA, false, false);
            accB = __builtin_amdgcn_wmma_f32_16x16x32_bf16(false, A1.v, false, Bf.v, (short)0, accB, false, false);
        }
        int col = nt * 16 + (lane & 15);
        #pragma unroll
        for (int r = 0; r < 8; ++r) {
            int m = r + sel * 8;                   // C layout: VGPR r -> row r / r+8
            if (m < ntok)      out[(size_t)toks[m]      * H + col] = accA[r];
            if (m + 16 < ntok) out[(size_t)toks[m + 16] * H + col] = accB[r];
        }
    }
}

// ---------------------------------------------------------------------------
extern "C" void kernel_launch(void* const* d_in, const int* in_sizes, int n_in,
                              void* d_out, int out_size, void* d_ws, size_t ws_size,
                              hipStream_t stream) {
    const float* x     = (const float*)d_in[0];
    const float* gatew = (const float*)d_in[1];
    const float* wg    = (const float*)d_in[2];
    const float* wu    = (const float*)d_in[3];
    const float* wd    = (const float*)d_in[4];
    float* out = (float*)d_out;

    constexpr int T = 16384, H = 1024, D = 256, E = 8;

    char* ws = (char*)d_ws;
    size_t off = 0;
    auto take = [&](size_t bytes) {
        char* p = ws + off;
        off = (off + bytes + 255) & ~(size_t)255;
        return p;
    };
    unsigned short* xb  = (unsigned short*)take((size_t)T * H * 2);   // 32 MB
    unsigned int*   wgp = (unsigned int*)  take((size_t)E * H * D * 2);
    unsigned int*   wup = (unsigned int*)  take((size_t)E * H * D * 2);
    unsigned int*   wdp = (unsigned int*)  take((size_t)E * D * H * 2);
    int*            bucket = (int*)take((size_t)E * T * 4);
    int*            counts = (int*)take(64);

    convert_x_kernel<<<4096, 256, 0, stream>>>(x, (uint2*)xb, counts, T * H / 4);
    pack_w_kernel<<<E * (H / 32) * (D / 16), 256, 0, stream>>>(wg, wgp, H, D);
    pack_w_kernel<<<E * (H / 32) * (D / 16), 256, 0, stream>>>(wu, wup, H, D);
    pack_w_kernel<<<E * (D / 32) * (H / 16), 256, 0, stream>>>(wd, wdp, D, H);
    router_kernel<<<T / 256, 256, 0, stream>>>(x, gatew, bucket, counts);

    dim3 grid(E, T / 32);
    moe_ffn_kernel<<<grid, 256, 0, stream>>>(xb, wgp, wup, wdp, bucket, counts, out);
}
